// GCN_52767968199068
// MI455X (gfx1250) — compile-verified, matrix operand order checked by reading, not run on hardware
//
#include <hip/hip_runtime.h>
#include <hip/hip_bf16.h>
#include <limits.h>

typedef __attribute__((ext_vector_type(16))) _Float16 v16h;
typedef __attribute__((ext_vector_type(8)))  _Float16 v8h;
typedef __attribute__((ext_vector_type(8)))  float    v8f;

#define EPS 1e-5f

// ---------- helpers ----------
__device__ __forceinline__ float wave_sum32(float x) {
    for (int off = 16; off > 0; off >>= 1) x += __shfl_xor(x, off, 32);
    return x;
}
// monotone float<->int key for atomicMax over signed floats
__device__ __forceinline__ int f2key(float f) {
    int b = __float_as_int(f);
    return (b >= 0) ? b : (b ^ 0x7fffffff);
}
__device__ __forceinline__ float key2f(int k) {
    int b = (k >= 0) ? k : (k ^ 0x7fffffff);
    return __int_as_float(b);
}

// ---------- init ----------
__global__ void fill_f32(float* __restrict__ p, float v, long long n) {
    long long i = (long long)blockIdx.x * blockDim.x + threadIdx.x;
    long long st = (long long)gridDim.x * blockDim.x;
    for (; i < n; i += st) p[i] = v;
}
__global__ void fill_i32(int* __restrict__ p, int v, long long n) {
    long long i = (long long)blockIdx.x * blockDim.x + threadIdx.x;
    long long st = (long long)gridDim.x * blockDim.x;
    for (; i < n; i += st) p[i] = v;
}

// ---------- GCN: x @ W^T for 3 linears fused (5 -> 64 each) ----------
__global__ void gcn_lin(const float* __restrict__ h,
                        const float* __restrict__ w_sum,
                        const float* __restrict__ w_mean,
                        const float* __restrict__ w_num,
                        float* __restrict__ xw, int N) {
    int idx = blockIdx.x * blockDim.x + threadIdx.x;
    if (idx >= N * 192) return;
    int i = idx / 192, j = idx % 192;
    const float* hp = h + (long long)i * 5;
    const float* w; int jj;
    if (j < 64)       { w = w_sum;  jj = j;       }
    else if (j < 128) { w = w_mean; jj = j - 64;  }
    else              { w = w_num;  jj = j - 128; }
    const float* wr = w + jj * 5;
    float s = 0.f;
#pragma unroll
    for (int k = 0; k < 5; ++k) s += hp[k] * wr[k];
    xw[idx] = s;
}

// ---------- GCN edge scatter (one wave32 per edge) ----------
__global__ void gcn_edge(const int* __restrict__ src, const int* __restrict__ dst,
                         const float* __restrict__ ew, const float* __restrict__ xw,
                         float* __restrict__ acc, float* __restrict__ deg,
                         float* __restrict__ sumw, int E) {
    int e = (blockIdx.x * blockDim.x + threadIdx.x) >> 5;
    if (e >= E) return;
    int lane = threadIdx.x & 31;
    int s = src[e], d = dst[e];
    float w = ew[e];
    const float* xp = xw + (long long)s * 192;
    float* ap = acc + (long long)d * 192;
#pragma unroll
    for (int t = 0; t < 6; ++t) {
        int j = lane + 32 * t;
        float v = xp[j];
        if (j < 128) v *= w;          // sum & mean convs use edge_weight, num conv does not
        atomicAdd(&ap[j], v);
    }
    if (lane == 0) { atomicAdd(&deg[d], 1.0f); atomicAdd(&sumw[d], w); }
}

// ---------- GCN node: bias + mean-div + 3x LayerNorm(64) -> hc (f32 + f16) ----------
__global__ void gcn_node(const float* __restrict__ acc, const float* __restrict__ deg,
                         const float* __restrict__ sumw,
                         const float* __restrict__ b_sum, const float* __restrict__ b_mean,
                         const float* __restrict__ b_num,
                         const float* __restrict__ g1, const float* __restrict__ bb1,
                         const float* __restrict__ g2, const float* __restrict__ bb2,
                         const float* __restrict__ g3, const float* __restrict__ bb3,
                         float* __restrict__ hc, _Float16* __restrict__ hch,
                         float* __restrict__ loop_attr, int N) {
    int i = (blockIdx.x * blockDim.x + threadIdx.x) >> 5;
    if (i >= N) return;
    int lane = threadIdx.x & 31;
    float dmax = fmaxf(deg[i], 1.0f);
    if (lane == 0) loop_attr[i] = sumw[i] / dmax;
    const float* biases[3] = { b_sum, b_mean, b_num };
    const float* gs[3] = { g1, g2, g3 };
    const float* bs[3] = { bb1, bb2, bb3 };
    const float* ap = acc + (long long)i * 192;
#pragma unroll
    for (int blk = 0; blk < 3; ++blk) {
        float x0 = ap[blk * 64 + lane];
        float x1 = ap[blk * 64 + lane + 32];
        if (blk == 1) { x0 /= dmax; x1 /= dmax; }
        x0 += biases[blk][lane];
        x1 += biases[blk][lane + 32];
        float m = wave_sum32(x0 + x1) * (1.0f / 64.0f);
        float d0 = x0 - m, d1 = x1 - m;
        float v = wave_sum32(d0 * d0 + d1 * d1) * (1.0f / 64.0f);
        float r = rsqrtf(v + EPS);
        float y0 = d0 * r * gs[blk][lane]      + bs[blk][lane];
        float y1 = d1 * r * gs[blk][lane + 32] + bs[blk][lane + 32];
        long long o = (long long)i * 192 + blk * 64;
        hc[o + lane] = y0;        hc[o + lane + 32] = y1;
        hch[o + lane] = (_Float16)y0; hch[o + lane + 32] = (_Float16)y1;
    }
}

// ---------- pack weight [384,192] into WMMA B fragments [6][24][512] halves ----------
__global__ void pack_wmma_b(const float* __restrict__ W, _Float16* __restrict__ Bf) {
    int idx = blockIdx.x * blockDim.x + threadIdx.x;
    if (idx >= 6 * 24 * 512) return;
    int tile = idx >> 9;
    int r = idx & 511;
    int lane = r >> 4, j = r & 15;
    int kt = tile / 24, nt = tile % 24;
    int k = kt * 32 + ((lane >> 4) * 8) + ((j >> 3) * 16) + (j & 7);
    int n = nt * 16 + (lane & 15);
    Bf[idx] = (_Float16)W[n * 192 + k];
}

// ---------- WMMA GEMM: C[M,384] = A[M,192](f16) x B + bias ----------
// One wave computes a 16x64 strip (4 N-tiles) so the A fragment is loaded once
// per K-step and reused from registers across 4 back-to-back WMMAs.
__global__ void gemm_wmma(const _Float16* __restrict__ A,
                          const _Float16* __restrict__ Bf,
                          const float* __restrict__ bias,
                          float* __restrict__ C) {
    int mt = blockIdx.x;            // M/16
    int ng = blockIdx.y;            // 0..5 (each covers 4 of 24 n-tiles)
    int lane = threadIdx.x;         // 0..31
    int m0 = mt * 16;
    int nt0 = ng * 4;
    int kb = (lane >> 4) * 8;
    long long arow = (long long)(m0 + (lane & 15)) * 192;
    v8f c0 = {}, c1 = {}, c2 = {}, c3 = {};
#pragma unroll
    for (int kt = 0; kt < 6; ++kt) {
        const _Float16* ap = A + arow + kt * 32 + kb;
        if (kt < 5) __builtin_prefetch(ap + 32, 0, 0);   // global_prefetch_b8
        v8h alo = *(const v8h*)(ap);
        v8h ahi = *(const v8h*)(ap + 16);
        v16h a = __builtin_shufflevector(alo, ahi,
                 0,1,2,3,4,5,6,7,8,9,10,11,12,13,14,15);
        const _Float16* bp = Bf + ((kt * 24 + nt0) * 512) + lane * 16;
        v16h b0 = *(const v16h*)(bp);
        v16h b1 = *(const v16h*)(bp + 512);
        v16h b2 = *(const v16h*)(bp + 1024);
        v16h b3 = *(const v16h*)(bp + 1536);
        c0 = __builtin_amdgcn_wmma_f32_16x16x32_f16(false, a, false, b0, (short)0, c0, false, false);
        c1 = __builtin_amdgcn_wmma_f32_16x16x32_f16(false, a, false, b1, (short)0, c1, false, false);
        c2 = __builtin_amdgcn_wmma_f32_16x16x32_f16(false, a, false, b2, (short)0, c2, false, false);
        c3 = __builtin_amdgcn_wmma_f32_16x16x32_f16(false, a, false, b3, (short)0, c3, false, false);
    }
    int nl = lane & 15;
    int mbase = m0 + ((lane >> 4) * 8);
    {
        int n = (nt0 + 0) * 16 + nl; float bn = bias[n];
#pragma unroll
        for (int i = 0; i < 8; ++i) C[(long long)(mbase + i) * 384 + n] = c0[i] + bn;
    }
    {
        int n = (nt0 + 1) * 16 + nl; float bn = bias[n];
#pragma unroll
        for (int i = 0; i < 8; ++i) C[(long long)(mbase + i) * 384 + n] = c1[i] + bn;
    }
    {
        int n = (nt0 + 2) * 16 + nl; float bn = bias[n];
#pragma unroll
        for (int i = 0; i < 8; ++i) C[(long long)(mbase + i) * 384 + n] = c2[i] + bn;
    }
    {
        int n = (nt0 + 3) * 16 + nl; float bn = bias[n];
#pragma unroll
        for (int i = 0; i < 8; ++i) C[(long long)(mbase + i) * 384 + n] = c3[i] + bn;
    }
}

// ---------- GATv2 attention logits + segment max (one wave per edge incl self loops) ----------
__global__ void gat_alpha(const int* __restrict__ src, const int* __restrict__ dst,
                          const float* __restrict__ ew, const float* __restrict__ loop_attr,
                          const float* __restrict__ xl, const float* __restrict__ xr,
                          const float* __restrict__ we, const float* __restrict__ att,
                          float* __restrict__ alphaE, int* __restrict__ mxI,
                          int E, int Etot) {
    int e = (blockIdx.x * blockDim.x + threadIdx.x) >> 5;
    if (e >= Etot) return;
    int lane = threadIdx.x & 31;
    int s, d; float ea;
    if (e < E) { s = src[e]; d = dst[e]; ea = ew[e]; }
    else       { s = e - E;  d = s;      ea = loop_attr[s]; }
    const float* xls = xl + (long long)s * 384;
    const float* xrd = xr + (long long)d * 384;
    float a0 = 0.f, a1 = 0.f;
#pragma unroll
    for (int t = 0; t < 12; ++t) {
        int j = lane + 32 * t;
        float z = xls[j] + xrd[j] + ea * we[j];
        z = (z > 0.f) ? z : 0.2f * z;        // leaky_relu 0.2
        float cterm = z * att[j];
        if (t < 6) a0 += cterm; else a1 += cterm;
    }
    a0 = wave_sum32(a0);
    a1 = wave_sum32(a1);
    if (lane == 0) {
        alphaE[(long long)e * 2]     = a0;
        alphaE[(long long)e * 2 + 1] = a1;
        atomicMax(&mxI[d * 2],     f2key(a0));
        atomicMax(&mxI[d * 2 + 1], f2key(a1));
    }
}

// ---------- GATv2 exp / denominator / unnormalized message accumulation ----------
__global__ void gat_accum(const int* __restrict__ src, const int* __restrict__ dst,
                          const float* __restrict__ alphaE, const int* __restrict__ mxI,
                          const float* __restrict__ xl,
                          float* __restrict__ den, float* __restrict__ msg,
                          int E, int Etot) {
    int e = (blockIdx.x * blockDim.x + threadIdx.x) >> 5;
    if (e >= Etot) return;
    int lane = threadIdx.x & 31;
    int s, d;
    if (e < E) { s = src[e]; d = dst[e]; }
    else       { s = e - E;  d = s; }
    float e0 = expf(alphaE[(long long)e * 2]     - key2f(mxI[d * 2]));
    float e1 = expf(alphaE[(long long)e * 2 + 1] - key2f(mxI[d * 2 + 1]));
    if (lane == 0) { atomicAdd(&den[d * 2], e0); atomicAdd(&den[d * 2 + 1], e1); }
    const float* xls = xl + (long long)s * 384;
    float* mp = msg + (long long)d * 384;
#pragma unroll
    for (int t = 0; t < 12; ++t) {
        int j = lane + 32 * t;
        atomicAdd(&mp[j], xls[j] * ((j < 192) ? e0 : e1));
    }
}

// ---------- final: softmax-normalize, head-mean, +bias, LN(192), LN(5) edge feats, FC(197->5) ----------
__global__ void final_node(const float* __restrict__ msg, const float* __restrict__ den,
                           const float* __restrict__ deg, const float* __restrict__ gat_b,
                           const float* __restrict__ lnA_g, const float* __restrict__ lnA_b,
                           const float* __restrict__ edge_num,
                           const float* __restrict__ lnE_g, const float* __restrict__ lnE_b,
                           const float* __restrict__ fc_w, const float* __restrict__ fc_b,
                           float* __restrict__ out, int N) {
    int i = (blockIdx.x * blockDim.x + threadIdx.x) >> 5;
    if (i >= N) return;
    int lane = threadIdx.x & 31;
    float inv_cnt2 = 1.0f / (deg[i] + 1.0f);       // self-loop always present
    float id0 = 1.0f / den[i * 2];
    float id1 = 1.0f / den[i * 2 + 1];
    const float* mp = msg + (long long)i * 384;
    float yv[6];
    float s = 0.f;
#pragma unroll
    for (int t = 0; t < 6; ++t) {
        int c = lane + 32 * t;
        float v = 0.5f * (mp[c] * id0 + mp[192 + c] * id1) * inv_cnt2 + gat_b[c];
        yv[t] = v; s += v;
    }
    float m = wave_sum32(s) * (1.0f / 192.0f);
    float vv = 0.f;
#pragma unroll
    for (int t = 0; t < 6; ++t) { float dd = yv[t] - m; vv += dd * dd; }
    vv = wave_sum32(vv) * (1.0f / 192.0f);
    float r = rsqrtf(vv + EPS);
#pragma unroll
    for (int t = 0; t < 6; ++t) {
        int c = lane + 32 * t;
        yv[t] = (yv[t] - m) * r * lnA_g[c] + lnA_b[c];
    }
    // LN over the 5 raw edge_num features (redundant per lane; trivial)
    float en[5]; float em = 0.f;
#pragma unroll
    for (int k = 0; k < 5; ++k) { en[k] = edge_num[(long long)i * 5 + k]; em += en[k]; }
    em *= 0.2f;
    float ev = 0.f;
#pragma unroll
    for (int k = 0; k < 5; ++k) { float dd = en[k] - em; ev += dd * dd; }
    ev *= 0.2f;
    float er = rsqrtf(ev + EPS);
#pragma unroll
    for (int k = 0; k < 5; ++k) en[k] = (en[k] - em) * er * lnE_g[k] + lnE_b[k];
    // FC: [197] -> [5]
#pragma unroll
    for (int o = 0; o < 5; ++o) {
        float p = 0.f;
#pragma unroll
        for (int t = 0; t < 6; ++t) {
            int c = lane + 32 * t;
            p += yv[t] * fc_w[o * 197 + c];
        }
        p = wave_sum32(p);
        if (lane == 0) {
            float q = p;
#pragma unroll
            for (int k = 0; k < 5; ++k) q += en[k] * fc_w[o * 197 + 192 + k];
            out[(long long)i * 5 + o] = q + fc_b[o];
        }
    }
}

// ---------------- host launch ----------------
extern "C" void kernel_launch(void* const* d_in, const int* in_sizes, int n_in,
                              void* d_out, int out_size, void* d_ws, size_t ws_size,
                              hipStream_t stream) {
    const float* h         = (const float*)d_in[0];
    const float* edge_num  = (const float*)d_in[1];
    const int*   edge_idx  = (const int*)  d_in[2];
    const float* edge_w    = (const float*)d_in[3];
    const float* w_sum     = (const float*)d_in[4];
    const float* b_sum     = (const float*)d_in[5];
    const float* w_mean    = (const float*)d_in[6];
    const float* b_mean    = (const float*)d_in[7];
    const float* w_num     = (const float*)d_in[8];
    const float* b_num     = (const float*)d_in[9];
    const float* ln1_g = (const float*)d_in[10]; const float* ln1_b = (const float*)d_in[11];
    const float* ln2_g = (const float*)d_in[12]; const float* ln2_b = (const float*)d_in[13];
    const float* ln3_g = (const float*)d_in[14]; const float* ln3_b = (const float*)d_in[15];
    const float* wl    = (const float*)d_in[16]; const float* bl    = (const float*)d_in[17];
    const float* wr    = (const float*)d_in[18]; const float* br    = (const float*)d_in[19];
    const float* we    = (const float*)d_in[20]; const float* att   = (const float*)d_in[21];
    const float* gat_b = (const float*)d_in[22];
    const float* lnA_g = (const float*)d_in[23]; const float* lnA_b = (const float*)d_in[24];
    const float* lnE_g = (const float*)d_in[25]; const float* lnE_b = (const float*)d_in[26];
    const float* fc_w  = (const float*)d_in[27]; const float* fc_b  = (const float*)d_in[28];

    const int N = in_sizes[0] / 5;
    const int E = in_sizes[3];
    const int Etot = E + N;
    const int* src = edge_idx;
    const int* dst = edge_idx + E;

    // workspace carve-up (floats), 256B-ish aligned chunks
    float* W = (float*)d_ws;
    size_t off = 0;
    auto alloc = [&](size_t nfloats) { size_t o = off; off += (nfloats + 63) & ~(size_t)63; return o; };
    float* msg  = W + alloc((size_t)N * 384);   // aliases xw (first half) + acc (second half)
    float* xw   = msg;
    float* acc  = msg + (size_t)N * 192;
    float* deg  = W + alloc(N);
    float* sumw = W + alloc(N);
    float* lpat = W + alloc(N);
    float* hc   = W + alloc((size_t)N * 192);   (void)hc;
    float* den  = W + alloc(2 * (size_t)N);
    float* alphaE = W + alloc(2 * (size_t)Etot);
    int*   mxI  = (int*)(W + alloc(2 * (size_t)N));
    _Float16* hch = (_Float16*)(W + alloc((size_t)N * 96));
    _Float16* Bl  = (_Float16*)(W + alloc(36864));
    _Float16* Br  = (_Float16*)(W + alloc(36864));
    float* xl   = W + alloc((size_t)N * 384);
    float* xr   = W + alloc((size_t)N * 384);

    dim3 b256(256);
    auto gs = [](long long n, int bs) { long long g = (n + bs - 1) / bs; return (unsigned)(g > 65535 ? 65535 : g); };

    // 1) zero accumulators (msg region covers xw+acc; xw fully overwritten below)
    fill_f32<<<gs((long long)N * 384, 256), b256, 0, stream>>>(msg, 0.f, (long long)N * 384);
    fill_f32<<<gs(N, 256), b256, 0, stream>>>(deg, 0.f, N);
    fill_f32<<<gs(N, 256), b256, 0, stream>>>(sumw, 0.f, N);
    fill_f32<<<gs(2LL * N, 256), b256, 0, stream>>>(den, 0.f, 2LL * N);
    fill_i32<<<gs(2LL * N, 256), b256, 0, stream>>>(mxI, INT_MIN, 2LL * N);

    // 2) GCN linears + weight packing for WMMA
    gcn_lin<<<gs((long long)N * 192, 256), b256, 0, stream>>>(h, w_sum, w_mean, w_num, xw, N);
    pack_wmma_b<<<gs(6 * 24 * 512, 256), b256, 0, stream>>>(wl, Bl);
    pack_wmma_b<<<gs(6 * 24 * 512, 256), b256, 0, stream>>>(wr, Br);

    // 3) GCN edge scatter + node LN -> hc (f32 + f16)
    gcn_edge<<<(E + 7) / 8, b256, 0, stream>>>(src, dst, edge_w, xw, acc, deg, sumw, E);
    gcn_node<<<(N + 7) / 8, b256, 0, stream>>>(acc, deg, sumw, b_sum, b_mean, b_num,
                                               ln1_g, ln1_b, ln2_g, ln2_b, ln3_g, ln3_b,
                                               hc, hch, lpat, N);

    // 4) re-zero msg region for GAT accumulation (GCN scratch now dead)
    fill_f32<<<gs((long long)N * 384, 256), b256, 0, stream>>>(msg, 0.f, (long long)N * 384);

    // 5) WMMA GEMMs: xl = hc @ wl^T + bl ; xr = hc @ wr^T + br
    dim3 ggrid((N + 15) / 16, 6);   // 4 n-tiles per wave
    gemm_wmma<<<ggrid, dim3(32), 0, stream>>>(hch, Bl, bl, xl);
    gemm_wmma<<<ggrid, dim3(32), 0, stream>>>(hch, Br, br, xr);

    // 6) GATv2 attention: logits+max, then exp/den/messages
    gat_alpha<<<(Etot + 7) / 8, b256, 0, stream>>>(src, dst, edge_w, lpat, xl, xr, we, att,
                                                   alphaE, mxI, E, Etot);
    gat_accum<<<(Etot + 7) / 8, b256, 0, stream>>>(src, dst, alphaE, mxI, xl, den, msg, E, Etot);

    // 7) finalize per node -> d_out [N,5]
    final_node<<<(N + 7) / 8, b256, 0, stream>>>(msg, den, deg, gat_b, lnA_g, lnA_b,
                                                 edge_num, lnE_g, lnE_b, fc_w, fc_b,
                                                 (float*)d_out, N);
    (void)n_in; (void)out_size; (void)ws_size;
}
